// Agent_12876311953937
// MI455X (gfx1250) — matrix operand order repaired and checked by reading, not compiled, not run on hardware
//
#include <hip/hip_runtime.h>
#include <hip/hip_bf16.h>
#include <math.h>

// ---------------------------------------------------------------------------
// MI455X (gfx1250) Mamba actor-critic forward.
// wave32, WGP. Dense math on V_WMMA_F32_16X16X32_F16 (f32 accumulate).
// u-tile staging in k2 via Tensor Data Mover (tensor_load_to_lds + TENSORcnt).
// Only the last token's y feeds the heads -> out_proj/einsum computed once.
// ---------------------------------------------------------------------------

#define B_DIM    32
#define L_DIM    1024
#define OBS_DIM  64
#define ACT_DIM  8
#define D_MODEL  128
#define D_INNER  256
#define D_STATE  16
#define DT_RANK  8
#define NTOK     (B_DIM * L_DIM)          // 32768 flat tokens
#define NTILE    (NTOK / 16)              // 2048 16-token tiles

static constexpr float LOG_2PI_F = 1.8378770664093453f;

typedef __attribute__((ext_vector_type(16))) _Float16     v16h;
typedef __attribute__((ext_vector_type(8)))  float        v8f;
typedef __attribute__((ext_vector_type(4)))  unsigned int v4u;
typedef __attribute__((ext_vector_type(8)))  int          v8i;
typedef __attribute__((ext_vector_type(4)))  int          v4i;

#if __has_builtin(__builtin_amdgcn_tensor_load_to_lds)
#define HAVE_TDM 1
#else
#define HAVE_TDM 0
#endif

// ------------------------- workspace layout (bytes) ------------------------
static constexpr size_t kAlign = 256;
static constexpr size_t au(size_t x) { return (x + kAlign - 1) / kAlign * kAlign; }
static constexpr size_t OFF_W1   = 0;                                          // 64x64   f16
static constexpr size_t OFF_W2   = au(OFF_W1  + 4096ull  * 2);                 // 64x64   f16
static constexpr size_t OFF_WP   = au(OFF_W2  + 4096ull  * 2);                 // 128x64  f16
static constexpr size_t OFF_INP  = au(OFF_WP  + 8192ull  * 2);                 // 512x128 f16
static constexpr size_t OFF_XP   = au(OFF_INP + 65536ull * 2);                 // 48x256  f16 (pad 40->48)
static constexpr size_t OFF_DTP  = au(OFF_XP  + 12288ull * 2);                 // 256x32  f16 (pad 8->32)
static constexpr size_t OFF_U    = au(OFF_DTP + 8192ull  * 2);                 // NTOK x 256 f32
static constexpr size_t OFF_DEL  = au(OFF_U   + (size_t)NTOK * 256 * 4);       // NTOK x 256 f32
static constexpr size_t OFF_DU   = au(OFF_DEL + (size_t)NTOK * 256 * 4);       // NTOK x 256 f32
static constexpr size_t OFF_BC   = au(OFF_DU  + (size_t)NTOK * 256 * 4);       // NTOK x 16 f32
static constexpr size_t OFF_CC   = au(OFF_BC  + (size_t)NTOK * 16  * 4);       // NTOK x 16 f32
static constexpr size_t OFF_RES  = au(OFF_CC  + (size_t)NTOK * 16  * 4);       // B x 256 f32
static constexpr size_t OFF_ULST = au(OFF_RES + 8192ull * 4);                  // B x 256 f32
static constexpr size_t OFF_YLST = au(OFF_ULST + 8192ull * 4);                 // B x 256 f32

// --------------------------- small device helpers ---------------------------
__device__ __forceinline__ void lds_wave_fence() {
  // cross-lane LDS producer->consumer within one wave: order + drain DScnt
  __builtin_amdgcn_wave_barrier();
  asm volatile("s_wait_dscnt 0" ::: "memory");
  __builtin_amdgcn_wave_barrier();
}

// Branch-free transcendentals: map onto v_exp_f32 / v_rcp_f32 TRANS ops that
// co-execute with the WMMA pipe (XDL-tracked) instead of libm's divergent
// save-exec polynomial paths.
__device__ __forceinline__ float fast_tanh(float x) {
  const float cx = fminf(fmaxf(x, -15.f), 15.f);
  const float e  = __expf(2.f * cx);
  return (e - 1.f) * __builtin_amdgcn_rcpf(e + 1.f);
}
__device__ __forceinline__ float fast_sigmoid(float x) {
  return __builtin_amdgcn_rcpf(1.f + __expf(-x));
}
__device__ __forceinline__ float fast_softplus(float x) {   // log(1+e^x), stable
  return fmaxf(x, 0.f) + __logf(1.f + __expf(-fabsf(x)));
}

// A fragment: 16x32 f16, row-major source [16][K] in LDS.
// Layout: lane l -> M = l&15; e=0..15 -> K = kc*32 + (l>>4)*16 + e (matches B).
__device__ __forceinline__ v16h load_a_frag(const _Float16* src, int K, int kc, int lane) {
  const int m = lane & 15, half = lane >> 4;
  const _Float16* p0 = src + m * K + kc * 32 + half * 16;
  v16h a;
#pragma unroll
  for (int e = 0; e < 8; ++e) { a[e] = p0[e]; a[e + 8] = p0[8 + e]; }
  return a;
}

// B fragment from row-major weight W[N][K] (f16):
// lane l -> N = pn*16 + (l&15); e -> K = kc*32 + (l>>4)*16 + e  (16 contiguous)
__device__ __forceinline__ v16h load_b_frag(const _Float16* W, int K, int pn, int kc, int lane) {
  const _Float16* p = W + (size_t)(pn * 16 + (lane & 15)) * K + kc * 32 + ((lane >> 4) << 4);
  return *reinterpret_cast<const v16h*>(p);  // 32B aligned -> b128 loads
}

// GEMM: LDS[16][K] f16  x  W[NP*16][K] f16  ->  LDS[16][NP*16] f16 (+bias, opt tanh)
template <int K, int NP, bool TANH>
__device__ __forceinline__ void gemm_act_lds(const _Float16* src, _Float16* dst,
                                             const _Float16* W, const float* bias, int lane) {
  v16h Af[K / 32];
#pragma unroll
  for (int kc = 0; kc < K / 32; ++kc) Af[kc] = load_a_frag(src, K, kc, lane);
  const int rb = (lane >> 4) << 3, c = lane & 15;
#pragma unroll
  for (int pn = 0; pn < NP; ++pn) {
    v8f acc = {};
#pragma unroll
    for (int kc = 0; kc < K / 32; ++kc)
      acc = __builtin_amdgcn_wmma_f32_16x16x32_f16(false, Af[kc], false,
                                                   load_b_frag(W, K, pn, kc, lane),
                                                   (short)0, acc, false, false);
    const float bv = bias[pn * 16 + c];
#pragma unroll
    for (int r = 0; r < 8; ++r) {
      float v = acc[r] + bv;
      if (TANH) v = fast_tanh(v);
      dst[(size_t)(rb + r) * (NP * 16) + pn * 16 + c] = (_Float16)v;
    }
  }
}

// ---------------- Tensor Data Mover: 2D f32 tile -> LDS (D# per ISA §8) -----
__device__ __forceinline__ void tdm_load_2d_f32(const float* gsrc, void* lds_dst,
                                                unsigned rows, unsigned cols,
                                                unsigned row_stride_elts) {
#if HAVE_TDM
  const unsigned long long ga = (unsigned long long)(size_t)gsrc;
  const unsigned lds = (unsigned)(size_t)lds_dst;
  v4u g0;
  g0[0] = 1u;                                             // count=1, user D#
  g0[1] = lds;                                            // lds_addr [63:32]
  g0[2] = (unsigned)(ga & 0xffffffffu);                   // global_addr [95:64]
  g0[3] = (unsigned)((ga >> 32) & 0x01ffffffu) | (2u << 30); // addr[56:32] | type=2
  v8i g1;
  g1[0] = (int)(2u << 16);                                // data_size=2 (4B), mask=0
  g1[1] = (int)((cols & 0xffffu) << 16);                  // tensor_dim0 lo16 @bit48
  g1[2] = (int)((cols >> 16) & 0xffffu)                   // tensor_dim0 hi16
        | (int)((rows & 0xffffu) << 16);                  // tensor_dim1 lo16 @bit80
  g1[3] = (int)((rows >> 16) & 0xffffu)                   // tensor_dim1 hi16
        | (int)((cols & 0xffffu) << 16);                  // tile_dim0 @bit112
  g1[4] = (int)(rows & 0xffffu);                          // tile_dim1 @bit128, tile_dim2=0
  g1[5] = (int)row_stride_elts;                           // tensor_dim0_stride lo32 @bit160
  g1[6] = 0;                                              // stride hi16 | dim1_stride lo16
  g1[7] = 0;
  const v4i z4 = {0, 0, 0, 0};
#if defined(__clang_major__) && __clang_major__ >= 23
  const v8i z8 = {0, 0, 0, 0, 0, 0, 0, 0};
  __builtin_amdgcn_tensor_load_to_lds(g0, g1, z4, z4, z8, 0);
#else
  __builtin_amdgcn_tensor_load_to_lds(g0, g1, z4, z4, 0);
#endif
#else
  (void)gsrc; (void)lds_dst; (void)rows; (void)cols; (void)row_stride_elts;
#endif
}

__device__ __forceinline__ void tdm_wait() {
#if HAVE_TDM
  __builtin_amdgcn_wave_barrier();
#if __has_builtin(__builtin_amdgcn_s_wait_tensorcnt)
  __builtin_amdgcn_s_wait_tensorcnt(0);
#else
  asm volatile("s_wait_tensorcnt 0" ::: "memory");
#endif
  asm volatile("" ::: "memory");
  __builtin_amdgcn_wave_barrier();
#endif
}

// ------------------------- k0: weight convert / pad -------------------------
__global__ __launch_bounds__(256) void k0_prep(const float* __restrict__ W1,
                                               const float* __restrict__ W2,
                                               const float* __restrict__ Wp,
                                               const float* __restrict__ inp,
                                               const float* __restrict__ xp,
                                               const float* __restrict__ dtp,
                                               _Float16* hW1, _Float16* hW2, _Float16* hWp,
                                               _Float16* hInp, _Float16* hXp, _Float16* hDtp) {
  int i = blockIdx.x * blockDim.x + threadIdx.x;
  if (i < 4096) hW1[i] = (_Float16)W1[i];
  else if (i < 8192) hW2[i - 4096] = (_Float16)W2[i - 4096];
  else if (i < 16384) hWp[i - 8192] = (_Float16)Wp[i - 8192];
  else if (i < 81920) hInp[i - 16384] = (_Float16)inp[i - 16384];
  else if (i < 94208) {               // x_proj: 40x256 -> 48x256 (zero rows)
    int j = i - 81920, n = j / 256, k = j % 256;
    hXp[j] = (_Float16)((n < 40) ? xp[n * 256 + k] : 0.f);
  } else if (i < 102400) {            // dt_proj: 256x8 -> 256x32 (zero cols)
    int j = i - 94208, n = j / 32, k = j % 32;
    hDtp[j] = (_Float16)((k < 8) ? dtp[n * 8 + k] : 0.f);
  }
}

// --------------- k1: MLP(tanh,tanh) -> Wp -> in_proj (all WMMA) -------------
__global__ __launch_bounds__(128) void k1_encoder_inproj(
    const float* __restrict__ x, const float* __restrict__ b1,
    const float* __restrict__ b2, const float* __restrict__ bp,
    const _Float16* __restrict__ hW1, const _Float16* __restrict__ hW2,
    const _Float16* __restrict__ hWp, const _Float16* __restrict__ hInp,
    float* __restrict__ u_ws, float* __restrict__ res_last) {
  __shared__ _Float16 smem[4][2][16 * 128];
  const int lane = threadIdx.x & 31, wv = threadIdx.x >> 5;
  const int tile = blockIdx.x * 4 + wv;
  const int t0 = tile * 16;
  _Float16* bufA = smem[wv][0];
  _Float16* bufB = smem[wv][1];

  // stage x tile (16 tokens x 64 obs) as f16
#pragma unroll
  for (int i = 0; i < 32; ++i) {
    int idx = i * 32 + lane;                 // idx = row*64 + col
    int row = idx >> 6, col = idx & 63;
    bufA[idx] = (_Float16)x[(size_t)(t0 + row) * OBS_DIM + col];
  }
  lds_wave_fence();

  gemm_act_lds<64, 4, true>(bufA, bufB, hW1, b1, lane);   lds_wave_fence();
  gemm_act_lds<64, 4, true>(bufB, bufA, hW2, b2, lane);   lds_wave_fence();
  gemm_act_lds<64, 8, false>(bufA, bufB, hWp, bp, lane);  lds_wave_fence();

  // in_proj: (16x128) x (128x512); n<256 -> u (all tokens), n>=256 -> res(last)
  v16h Af[4];
#pragma unroll
  for (int kc = 0; kc < 4; ++kc) Af[kc] = load_a_frag(bufB, 128, kc, lane);
  const int rb = (lane >> 4) << 3, c = lane & 15;
  for (int pn = 0; pn < 32; ++pn) {
    v8f acc = {};
#pragma unroll
    for (int kc = 0; kc < 4; ++kc)
      acc = __builtin_amdgcn_wmma_f32_16x16x32_f16(false, Af[kc], false,
                                                   load_b_frag(hInp, 128, pn, kc, lane),
                                                   (short)0, acc, false, false);
    const int n = pn * 16 + c;
#pragma unroll
    for (int r = 0; r < 8; ++r) {
      const int tok = t0 + rb + r;
      const float v = acc[r];
      if (n < D_INNER) u_ws[(size_t)tok * D_INNER + n] = v;
      else if ((tok & (L_DIM - 1)) == L_DIM - 1)
        res_last[(tok >> 10) * D_INNER + (n - D_INNER)] = v;
    }
  }
}

// -------- k2: TDM-staged causal conv4 + SiLU -> x_proj -> dt_proj (WMMA) ----
__global__ __launch_bounds__(64) void k2_conv_proj(
    const float* __restrict__ conv_w, const float* __restrict__ conv_b,
    const float* __restrict__ dt_proj_b,
    const _Float16* __restrict__ hXp, const _Float16* __restrict__ hDtp,
    const float* __restrict__ u_ws,
    float* __restrict__ delta_ws, float* __restrict__ du_ws,
    float* __restrict__ Bc_ws, float* __restrict__ Cc_ws,
    float* __restrict__ u_last) {
  __shared__ float    sUF[2][16 * 256];   // TDM-staged raw u tile (f32)
  __shared__ _Float16 sU[2][16 * 256];    // conv+silu output (f16, WMMA A)
  __shared__ _Float16 sDT[2][16 * 32];    // dt-rank panel (K padded to 32)
  const int lane = threadIdx.x & 31, wv = threadIdx.x >> 5;
  const int tile = blockIdx.x * 2 + wv;
  const int t0 = tile * 16;
  float*    UF = sUF[wv];
  _Float16* U  = sU[wv];
  _Float16* DT = sDT[wv];

  // Async DMA of this wave's 16x256 f32 u tile into LDS (TENSORcnt-tracked).
  tdm_load_2d_f32(u_ws + (size_t)t0 * D_INNER, UF, 16, 256, 256);

#pragma unroll
  for (int i = 0; i < 16; ++i) DT[i * 32 + lane] = (_Float16)0.f;  // K-pad zeros

  tdm_wait();

#if !HAVE_TDM
  // fallback staging if the TDM builtin is unavailable on this toolchain
  for (int i = 0; i < 128; ++i) {
    int idx = i * 32 + lane;
    UF[idx] = u_ws[(size_t)t0 * D_INNER + idx];
  }
  lds_wave_fence();
#endif

  // depthwise causal conv (taps 4) + SiLU; halo rows (3) come from global
  for (int i = 0; i < 128; ++i) {
    int idx = i * 32 + lane;                 // idx = row*256 + col
    int row = idx >> 8, col = idx & 255;
    int tok = t0 + row, pos = tok & (L_DIM - 1);
    float acc = conv_b[col];
#pragma unroll
    for (int j = 0; j < 4; ++j) {
      int rr = row - 3 + j;
      int pp = pos - 3 + j;
      if (pp >= 0) {
        float uv = (rr >= 0) ? UF[rr * 256 + col]
                             : u_ws[(size_t)(tok - 3 + j) * D_INNER + col];
        acc += conv_w[col * 4 + j] * uv;
      }
    }
    float s = acc * fast_sigmoid(acc);       // silu
    U[idx] = (_Float16)s;
    if (pos == L_DIM - 1) u_last[(tok >> 10) * D_INNER + col] = s;
  }
  lds_wave_fence();

  // x_proj: (16x256) x (256x48) -> dt(0:8) | B(8:24) | C(24:40)
  v16h Af[8];
#pragma unroll
  for (int kc = 0; kc < 8; ++kc) Af[kc] = load_a_frag(U, 256, kc, lane);
  const int rb = (lane >> 4) << 3, c = lane & 15;
#pragma unroll
  for (int pn = 0; pn < 3; ++pn) {
    v8f acc = {};
#pragma unroll
    for (int kc = 0; kc < 8; ++kc)
      acc = __builtin_amdgcn_wmma_f32_16x16x32_f16(false, Af[kc], false,
                                                   load_b_frag(hXp, 256, pn, kc, lane),
                                                   (short)0, acc, false, false);
    const int col = pn * 16 + c;
#pragma unroll
    for (int r = 0; r < 8; ++r) {
      const int tr = rb + r, tok = t0 + tr;
      const float v = acc[r];
      if (col < 8)       DT[tr * 32 + col] = (_Float16)v;
      else if (col < 24) Bc_ws[(size_t)tok * D_STATE + (col - 8)] = v;
      else if (col < 40) Cc_ws[(size_t)tok * D_STATE + (col - 24)] = v;
    }
  }
  lds_wave_fence();

  // dt_proj: (16x32pad) x (32x256), softplus; also emit du = delta*u
  const v16h Ad = load_a_frag(DT, 32, 0, lane);
  for (int pn = 0; pn < 16; ++pn) {
    v8f acc = {};
    acc = __builtin_amdgcn_wmma_f32_16x16x32_f16(false, Ad, false,
                                                 load_b_frag(hDtp, 32, pn, 0, lane),
                                                 (short)0, acc, false, false);
    const int n = pn * 16 + c;
    const float bias = dt_proj_b[n];
#pragma unroll
    for (int r = 0; r < 8; ++r) {
      const int tr = rb + r, tok = t0 + tr;
      const float delta = fast_softplus(acc[r] + bias);
      const float uc = (float)U[tr * 256 + n];
      delta_ws[(size_t)tok * D_INNER + n] = delta;
      du_ws[(size_t)tok * D_INNER + n]    = delta * uc;
    }
  }
}

// ---------------- k3: sequential selective scan (1 block / batch) -----------
__global__ __launch_bounds__(256) void k3_scan(
    const float* __restrict__ delta_ws, const float* __restrict__ du_ws,
    const float* __restrict__ Bc_ws, const float* __restrict__ Cc_ws,
    const float* __restrict__ A_log, const float* __restrict__ Dskip,
    const float* __restrict__ u_last, const float* __restrict__ res_last,
    float* __restrict__ y_last) {
  const int b = blockIdx.x, d = threadIdx.x;  // d in [0,256)
  __shared__ float sB[D_STATE];

  float a[D_STATE], h[D_STATE];
#pragma unroll
  for (int n = 0; n < D_STATE; ++n) { a[n] = -__expf(A_log[d * D_STATE + n]); h[n] = 0.f; }

  const size_t base = (size_t)b * L_DIM;
  const float* dp  = delta_ws + base * D_INNER + d;
  const float* dup = du_ws    + base * D_INNER + d;

  for (int t = 0; t < L_DIM; ++t) {
    if (threadIdx.x < D_STATE) sB[threadIdx.x] = Bc_ws[(base + t) * D_STATE + threadIdx.x];
    __syncthreads();
    const float de  = dp[(size_t)t * D_INNER];
    const float duv = dup[(size_t)t * D_INNER];
    __builtin_prefetch(dp  + (size_t)(t + 8) * D_INNER, 0, 1);
    __builtin_prefetch(dup + (size_t)(t + 8) * D_INNER, 0, 1);
#pragma unroll
    for (int n = 0; n < D_STATE; ++n)
      h[n] = __expf(de * a[n]) * h[n] + duv * sB[n];
    __syncthreads();
  }

  // y at last token only: <h, C> + u*D, gated by silu(res)
  float y = 0.f;
#pragma unroll
  for (int n = 0; n < D_STATE; ++n)
    y += h[n] * Cc_ws[(base + L_DIM - 1) * D_STATE + n];
  y += u_last[b * D_INNER + d] * Dskip[d];
  const float r = res_last[b * D_INNER + d];
  y *= r * fast_sigmoid(r);
  y_last[b * D_INNER + d] = y;
}

// ----------------------------- k4: tiny heads -------------------------------
__global__ __launch_bounds__(128) void k4_heads(
    const float* __restrict__ y_last, const float* __restrict__ out_proj_w,
    const float* __restrict__ Wc, const float* __restrict__ bc,
    const float* __restrict__ Wa, const float* __restrict__ ba,
    const float* __restrict__ actor_logstd, const float* __restrict__ action,
    float* __restrict__ out) {
  const int b = blockIdx.x, j = threadIdx.x;  // j in [0,128)
  __shared__ float sLast[D_MODEL];
  __shared__ float sLP[ACT_DIM];

  float acc = 0.f;
  for (int k = 0; k < D_INNER; ++k) acc += y_last[b * D_INNER + k] * out_proj_w[(size_t)j * D_INNER + k];
  sLast[j] = acc;
  __syncthreads();

  if (j < ACT_DIM) {
    float m = ba[j];
    for (int k = 0; k < D_MODEL; ++k) m += sLast[k] * Wa[j * D_MODEL + k];
    const float av = action[b * ACT_DIM + j];
    out[b * ACT_DIM + j] = av;               // action passthrough
    const float ls = actor_logstd[j];
    const float sd = __expf(ls);
    const float diff = av - m;
    sLP[j] = -diff * diff / (2.f * sd * sd) - ls - 0.5f * LOG_2PI_F;
  }
  __syncthreads();

  if (j == 0) {
    float lp = 0.f, ent = 0.f;
    for (int a2 = 0; a2 < ACT_DIM; ++a2) { lp += sLP[a2]; ent += 0.5f + 0.5f * LOG_2PI_F + actor_logstd[a2]; }
    out[256 + b] = lp;        // logprob
    out[288 + b] = ent;       // entropy
    float v = bc[0];
    for (int k = 0; k < D_MODEL; ++k) v += sLast[k] * Wc[k];
    out[320 + b] = v;         // value
  }
}

// ------------------------------- launcher -----------------------------------
extern "C" void kernel_launch(void* const* d_in, const int* in_sizes, int n_in,
                              void* d_out, int out_size, void* d_ws, size_t ws_size,
                              hipStream_t stream) {
  (void)in_sizes; (void)n_in; (void)out_size; (void)ws_size;
  const float* x        = (const float*)d_in[0];
  const float* action   = (const float*)d_in[1];
  const float* W1       = (const float*)d_in[2];
  const float* b1       = (const float*)d_in[3];
  const float* W2       = (const float*)d_in[4];
  const float* b2       = (const float*)d_in[5];
  const float* Wp       = (const float*)d_in[6];
  const float* bp       = (const float*)d_in[7];
  const float* in_proj  = (const float*)d_in[8];
  const float* conv_w   = (const float*)d_in[9];
  const float* conv_b   = (const float*)d_in[10];
  const float* x_proj   = (const float*)d_in[11];
  const float* dt_projw = (const float*)d_in[12];
  const float* dt_projb = (const float*)d_in[13];
  const float* A_log    = (const float*)d_in[14];
  const float* Dskip    = (const float*)d_in[15];
  const float* out_proj = (const float*)d_in[16];
  const float* Wc       = (const float*)d_in[17];
  const float* bc       = (const float*)d_in[18];
  const float* Wa       = (const float*)d_in[19];
  const float* ba       = (const float*)d_in[20];
  const float* logstd   = (const float*)d_in[21];

  char* ws = (char*)d_ws;
  _Float16* hW1  = (_Float16*)(ws + OFF_W1);
  _Float16* hW2  = (_Float16*)(ws + OFF_W2);
  _Float16* hWp  = (_Float16*)(ws + OFF_WP);
  _Float16* hInp = (_Float16*)(ws + OFF_INP);
  _Float16* hXp  = (_Float16*)(ws + OFF_XP);
  _Float16* hDtp = (_Float16*)(ws + OFF_DTP);
  float* u_ws     = (float*)(ws + OFF_U);
  float* delta_ws = (float*)(ws + OFF_DEL);
  float* du_ws    = (float*)(ws + OFF_DU);
  float* Bc_ws    = (float*)(ws + OFF_BC);
  float* Cc_ws    = (float*)(ws + OFF_CC);
  float* res_last = (float*)(ws + OFF_RES);
  float* u_last   = (float*)(ws + OFF_ULST);
  float* y_last   = (float*)(ws + OFF_YLST);

  k0_prep<<<400, 256, 0, stream>>>(W1, W2, Wp, in_proj, x_proj, dt_projw,
                                   hW1, hW2, hWp, hInp, hXp, hDtp);
  k1_encoder_inproj<<<NTILE / 4, 128, 0, stream>>>(x, b1, b2, bp, hW1, hW2, hWp, hInp,
                                                   u_ws, res_last);
  k2_conv_proj<<<NTILE / 2, 64, 0, stream>>>(conv_w, conv_b, dt_projb, hXp, hDtp, u_ws,
                                             delta_ws, du_ws, Bc_ws, Cc_ws, u_last);
  k3_scan<<<B_DIM, 256, 0, stream>>>(delta_ws, du_ws, Bc_ws, Cc_ws, A_log, Dskip,
                                     u_last, res_last, y_last);
  k4_heads<<<B_DIM, 128, 0, stream>>>(y_last, out_proj, Wc, bc, Wa, ba, logstd, action,
                                      (float*)d_out);
}
